// LocalAttn_14826227106231
// MI455X (gfx1250) — matrix operand order repaired and checked by reading, not compiled, not run on hardware
//
#include <hip/hip_runtime.h>
#include <hip/hip_bf16.h>

typedef _Float16 f16;
typedef __attribute__((ext_vector_type(16))) _Float16 v16h;
typedef __attribute__((ext_vector_type(8)))  _Float16 v8h;
typedef __attribute__((ext_vector_type(2)))  _Float16 h2;
typedef __attribute__((ext_vector_type(8)))  float    v8f;

#define B_ 8
#define S_ 1024
#define E_ 512
#define H_ 16
#define D_ 32
#define L_ 1024

static __device__ __forceinline__ v8f wmma_f16(v16h a, v16h b, v8f c) {
  // 8 args: (neg_a, A, neg_b, B, c_mod, C, reuse_a, reuse_b)
  return __builtin_amdgcn_wmma_f32_16x16x32_f16(false, a, false, b, (short)0, c,
                                                false, false);
}

// ---------------------------------------------------------------------------
// Kernel 0: bulk f32 -> f16 conversion (8 elements / lane)
// ---------------------------------------------------------------------------
__global__ void cvt_f32_f16_kernel(const float* __restrict__ src,
                                   f16* __restrict__ dst, int n8) {
  const int i = blockIdx.x * blockDim.x + threadIdx.x;
  if (i >= n8) return;
  v8f s = *reinterpret_cast<const v8f*>(src + (size_t)i * 8);
  v8h d;
#pragma unroll
  for (int e = 0; e < 8; ++e) d[e] = (f16)s[e];
  *reinterpret_cast<v8h*>(dst + (size_t)i * 8) = d;
}

// ---------------------------------------------------------------------------
// Kernel 1: QKV projection  y = x @ W^T   (M=B*S=8192, N=E=512, K=E=512)
// All-f16 operands, 64x32 tile per wave, 8 WMMAs per k-step, spill-free.
// The two B tiles cover even/odd interleaved columns (n = n0 + 2c + bt) so
// each lane holds results for adjacent d -> packed b32 stores.
// Block-uniform batch index (m0 multiple of 64) collapses store addressing.
// Q,K stored f16 [B,H,S,D]; V stored f16 transposed [B,H,D,S].
// ---------------------------------------------------------------------------
__global__ void qkv_gemm_kernel(const f16* __restrict__ x16,
                                const f16* __restrict__ W16,   // [3][E][E]
                                f16* __restrict__ Qws,
                                f16* __restrict__ Kws,
                                f16* __restrict__ Vtws) {
  const int lane = threadIdx.x;
  const int c = lane & 15;     // tile col (B/C) or tile row (A)
  const int g = lane >> 4;     // lane-half -> K sub-range
  const int n0 = blockIdx.x * 32;
  const int m0 = blockIdx.y * 64;
  const int which = blockIdx.z;
  const f16* __restrict__ W = W16 + (size_t)which * E_ * E_;

  v8f acc[4][2] = {};

  const f16* ap0 = x16 + (size_t)(m0 + c) * E_;
  const f16* bp0 = W + (size_t)(n0 + 2 * c) * E_ + g * 16;   // even/odd pair base

  for (int k0 = 0; k0 < E_; k0 += 32) {
    v16h a[4], bm[2];
#pragma unroll
    for (int t = 0; t < 4; ++t) {
      // A operand: lane m holds K = g*16+e
      const f16* ap = ap0 + (size_t)(t * 16) * E_ + k0;
      v8h a0 = *reinterpret_cast<const v8h*>(ap + g * 16);
      v8h a1 = *reinterpret_cast<const v8h*>(ap + g * 16 + 8);
#pragma unroll
      for (int e = 0; e < 8; ++e) { a[t][e] = a0[e]; a[t][8 + e] = a1[e]; }
    }
#pragma unroll
    for (int t = 0; t < 2; ++t) {
      // B operand column c is W row n0 + 2c + t; K = g*16 + e
      const f16* bp = bp0 + (size_t)t * E_ + k0;
      bm[t] = *reinterpret_cast<const v16h*>(bp);
    }
#pragma unroll
    for (int at = 0; at < 4; ++at)
#pragma unroll
      for (int bt = 0; bt < 2; ++bt)
        acc[at][bt] = wmma_f16(a[at], bm[bt], acc[at][bt]);
  }

  // ---- write-out. C layout: VGPR v, lanes 0-15 -> M=v, lanes 16-31 -> M=v+8.
  // C column c corresponds to output feature n = n0 + 2c + bt -> d = 2c + bt.
  // m0 is 64-aligned -> batch index uniform across the tile.
  const int h  = n0 >> 5;                 // head index (block-uniform)
  const int bb = m0 >> 10;                // batch index (block-uniform)
  const size_t bh = (size_t)bb * H_ + h;
  const int sbase = (m0 & (S_ - 1)) + 8 * g;

  if (which != 2) {
    f16* __restrict__ dst = (which == 0 ? Qws : Kws) + bh * S_ * D_ + 2 * c;
#pragma unroll
    for (int at = 0; at < 4; ++at) {
#pragma unroll
      for (int v = 0; v < 8; ++v) {
        const int s = sbase + at * 16 + v;
        h2 pv = { (f16)acc[at][0][v], (f16)acc[at][1][v] };   // d = 2c, 2c+1
        *reinterpret_cast<h2*>(dst + (size_t)s * D_) = pv;
      }
    }
  } else {
#pragma unroll
    for (int bt = 0; bt < 2; ++bt) {
      const int d = 2 * c + bt;
      f16* __restrict__ dst = Vtws + (bh * D_ + d) * S_ + sbase;
#pragma unroll
      for (int at = 0; at < 4; ++at) {
#pragma unroll
        for (int v = 0; v < 8; v += 2) {                      // s, s+1 adjacent
          h2 pv = { (f16)acc[at][bt][v], (f16)acc[at][bt][v + 1] };
          *reinterpret_cast<h2*>(dst + at * 16 + v) = pv;
        }
      }
    }
  }
}

// ---------------------------------------------------------------------------
// Kernel 2: flash-style attention with relative-position bias.
// One wave per (b, h, 16-row query tile). Per 32-key block:
//   2x WMMA  S = Q K^T          (16x32 scores)
//   3x WMMA  R = Q Er^T         (16x48 window covering all diagonals)
//   diagonal gather Srel[r,c] = R[r][c+15-r] from LDS (conflict-free)
//   online softmax, P -> LDS -> A layout
//   2x WMMA  O += P V           (16x32 f32 accumulator)
// ---------------------------------------------------------------------------
__global__ void local_attn_kernel(const f16* __restrict__ Qws,
                                  const f16* __restrict__ Kws,
                                  const f16* __restrict__ Vtws,
                                  const f16* __restrict__ Er16,
                                  float* __restrict__ out) {
  __shared__ __align__(64) float Rld[16][48];
  __shared__ __align__(64) f16   Pld[16][32];

  const int lane = threadIdx.x;
  const int c = lane & 15;
  const int g = lane >> 4;
  const int i0 = blockIdx.x * 16;
  const int h  = blockIdx.y;
  const int b  = blockIdx.z;
  const size_t bh = (size_t)b * H_ + h;
  const float scale = 0.17677669529663687f;  // 1/sqrt(32)

  // Q tile as A operand (register-resident for all 7 WMMAs per key block)
  const f16* qrow = Qws + (bh * S_ + i0 + c) * D_;
  v16h qa;
  {
    v8h q0 = *reinterpret_cast<const v8h*>(qrow + g * 16);
    v8h q1 = *reinterpret_cast<const v8h*>(qrow + g * 16 + 8);
#pragma unroll
    for (int e = 0; e < 8; ++e) { qa[e] = q0[e]; qa[8 + e] = q1[e]; }
  }

  v8f m_run, l_run = {}, o0 = {}, o1 = {};
#pragma unroll
  for (int v = 0; v < 8; ++v) m_run[v] = -3.0e38f;

  for (int jb = 0; jb < S_; jb += 32) {
    // ---- S = Q K^T : two 16-column tiles
    v8f s0 = {}, s1 = {};
    {
      const f16* krow = Kws + (bh * S_ + jb + c) * D_ + g * 16;
      v16h kb0 = *reinterpret_cast<const v16h*>(krow);
      v16h kb1 = *reinterpret_cast<const v16h*>(krow + 16 * D_);
      __builtin_prefetch(krow + 32 * D_, 0, 1);   // next key block
      s0 = wmma_f16(qa, kb0, s0);
      s1 = wmma_f16(qa, kb1, s1);
    }

    // ---- R = Q Er^T over the diagonal window: p = p0 + [0,48)
    const int p0 = jb - i0 + (L_ - 16);
#pragma unroll
    for (int t = 0; t < 3; ++t) {
      int p = p0 + t * 16 + c;
      p = p < 0 ? 0 : (p > 2 * L_ - 2 ? 2 * L_ - 2 : p);
      const f16* erow = Er16 + (size_t)p * D_ + g * 16;
      v16h eb = *reinterpret_cast<const v16h*>(erow);
      v8f rz = {};
      rz = wmma_f16(qa, eb, rz);
#pragma unroll
      for (int v = 0; v < 8; ++v) Rld[v + 8 * g][t * 16 + c] = rz[v];
    }
    __syncthreads();

    // ---- scores = (QK + Srel) * scale ; Srel[r,c] = R[r][c + 15 - r]
#pragma unroll
    for (int v = 0; v < 8; ++v) {
      const int r = v + 8 * g;
      s0[v] = (s0[v] + Rld[r][c + 15 - r]) * scale;
      s1[v] = (s1[v] + Rld[r][c + 31 - r]) * scale;
    }
    __syncthreads();

    // ---- online softmax (rows live across 16-lane halves -> shfl width 16)
    v8f mx;
#pragma unroll
    for (int v = 0; v < 8; ++v) mx[v] = fmaxf(s0[v], s1[v]);
#pragma unroll
    for (int off = 8; off >= 1; off >>= 1)
#pragma unroll
      for (int v = 0; v < 8; ++v)
        mx[v] = fmaxf(mx[v], __shfl_xor(mx[v], off, 16));

    v8f mn, alpha;
#pragma unroll
    for (int v = 0; v < 8; ++v) {
      mn[v]    = fmaxf(m_run[v], mx[v]);
      alpha[v] = __expf(m_run[v] - mn[v]);
    }
#pragma unroll
    for (int v = 0; v < 8; ++v) {
      s0[v] = __expf(s0[v] - mn[v]);
      s1[v] = __expf(s1[v] - mn[v]);
    }
    v8f rs;
#pragma unroll
    for (int v = 0; v < 8; ++v) rs[v] = s0[v] + s1[v];
#pragma unroll
    for (int off = 8; off >= 1; off >>= 1)
#pragma unroll
      for (int v = 0; v < 8; ++v) rs[v] += __shfl_xor(rs[v], off, 16);
#pragma unroll
    for (int v = 0; v < 8; ++v) {
      l_run[v] = l_run[v] * alpha[v] + rs[v];
      m_run[v] = mn[v];
      o0[v] *= alpha[v];
      o1[v] *= alpha[v];
    }

    // ---- P (C layout) -> LDS f16 -> reload in A layout
#pragma unroll
    for (int v = 0; v < 8; ++v) {
      const int r = v + 8 * g;
      Pld[r][c]      = (f16)s0[v];
      Pld[r][16 + c] = (f16)s1[v];
    }
    __syncthreads();
    v16h pa;
    {
      v8h pav0 = *reinterpret_cast<const v8h*>(&Pld[c][g * 16]);
      v8h pav1 = *reinterpret_cast<const v8h*>(&Pld[c][g * 16 + 8]);
#pragma unroll
      for (int e = 0; e < 8; ++e) { pa[e] = pav0[e]; pa[8 + e] = pav1[e]; }
    }
    __syncthreads();

    // ---- O += P @ V : B operand from transposed V, contiguous per lane
    const f16* vrow0 = Vtws + (bh * D_ + c) * S_ + jb + g * 16;
    const f16* vrow1 = Vtws + (bh * D_ + 16 + c) * S_ + jb + g * 16;
    v16h vb0 = *reinterpret_cast<const v16h*>(vrow0);
    v16h vb1 = *reinterpret_cast<const v16h*>(vrow1);
    o0 = wmma_f16(pa, vb0, o0);
    o1 = wmma_f16(pa, vb1, o1);
  }

  // ---- normalize and store: out[b, s, h*32 + d]
#pragma unroll
  for (int v = 0; v < 8; ++v) {
    const int s = i0 + v + 8 * g;
    const float inv = 1.0f / l_run[v];
    float* orow = out + ((size_t)b * S_ + s) * E_ + h * D_;
    orow[c]      = o0[v] * inv;
    orow[16 + c] = o1[v] * inv;
  }
}

extern "C" void kernel_launch(void* const* d_in, const int* in_sizes, int n_in,
                              void* d_out, int out_size, void* d_ws, size_t ws_size,
                              hipStream_t stream) {
  const float* x  = (const float*)d_in[0];
  const float* Wq = (const float*)d_in[1];
  const float* Wk = (const float*)d_in[2];
  const float* Wv = (const float*)d_in[3];
  const float* Er = (const float*)d_in[4];
  float* out = (float*)d_out;

  const size_t BHSD = (size_t)B_ * H_ * S_ * D_;   // 4,194,304 elems
  const size_t WSZ  = (size_t)E_ * E_;             //   262,144 elems
  const size_t ERSZ = (size_t)(2 * L_ - 1) * D_;   //    65,504 elems

  f16* Qws  = (f16*)d_ws;            // [B,H,S,D]
  f16* Kws  = Qws + BHSD;            // [B,H,S,D]
  f16* Vt   = Kws + BHSD;            // [B,H,D,S] (transposed)
  f16* x16  = Vt + BHSD;             // [B*S, E]
  f16* W16  = x16 + BHSD;            // [3, E, E]
  f16* Er16 = W16 + 3 * WSZ;         // [2L-1, D]
  (void)in_sizes; (void)n_in; (void)out_size; (void)ws_size;

  // Stage 0: one-shot f32 -> f16 conversions (all sizes divisible by 8)
  {
    const int n8x = (int)(BHSD / 8), n8w = (int)(WSZ / 8), n8e = (int)(ERSZ / 8);
    cvt_f32_f16_kernel<<<(n8x + 255) / 256, 256, 0, stream>>>(x,  x16, n8x);
    cvt_f32_f16_kernel<<<(n8w + 255) / 256, 256, 0, stream>>>(Wq, W16,           n8w);
    cvt_f32_f16_kernel<<<(n8w + 255) / 256, 256, 0, stream>>>(Wk, W16 + WSZ,     n8w);
    cvt_f32_f16_kernel<<<(n8w + 255) / 256, 256, 0, stream>>>(Wv, W16 + 2 * WSZ, n8w);
    cvt_f32_f16_kernel<<<(n8e + 255) / 256, 256, 0, stream>>>(Er, Er16, n8e);
  }

  // Stage 1: QKV projections, 64x32 tile per wave (spill-free)
  dim3 gp(E_ / 32, (B_ * S_) / 64, 3);             // 16 x 128 x 3
  qkv_gemm_kernel<<<gp, 32, 0, stream>>>(x16, W16, Qws, Kws, Vt);

  // Stage 2: attention, one wave per (b, h, 16-query tile)
  dim3 ga(S_ / 16, H_, B_);                        // 64 x 16 x 8
  local_attn_kernel<<<ga, 32, 0, stream>>>(Qws, Kws, Vt, Er16, out);
}
